// MyNetwork_69801808495352
// MI455X (gfx1250) — matrix operand order repaired and checked by reading, not compiled, not run on hardware
//
#include <hip/hip_runtime.h>
#include <stdint.h>

#define N_ROWS   2048
#define D_IN     4096
#define H1       1024
#define HP       1025   // H1 + 1 (bias row of the RLS state)
#define D_OUT    16

typedef __attribute__((ext_vector_type(16))) __bf16 v16bf;
typedef __attribute__((ext_vector_type(8)))  float  v8f;
typedef unsigned int v4u  __attribute__((ext_vector_type(4)));
typedef int          v8i_t __attribute__((ext_vector_type(8)));
typedef int          v4i_t __attribute__((ext_vector_type(4)));

// ---------------------------------------------------------------------------
// Kernel 1: feature = relu(x @ W1^T + b1)
// fp32 -> bf16x3 split-precision WMMA (hi*hi + hi*lo + lo*hi), fp32 accum.
// Tiles staged into LDS by the Tensor Data Mover (tensor_load_to_lds),
// double-buffered, synchronized with s_wait_tensorcnt + s_barrier.
// Block = 256 threads (8 waves); block tile = 128(M) x 64(N);
// each wave computes a 16(M) x 64(N) strip.
// ---------------------------------------------------------------------------

#define KSTEP  32
#define XROWS  128
#define WROWS  64
#define PADW   36   // LDS row stride in dwords: 32 data + 4 pad (TDM pad_enable)

__device__ __forceinline__ float4 ld4(const float* p) { return *(const float4*)p; }
__device__ __forceinline__ void st4f(float* d, float4 a) {
  d[0] = a.x; d[1] = a.y; d[2] = a.z; d[3] = a.w;
}

// Low 32 bits of a generic LDS address are the LDS byte offset (ISA 10.2).
__device__ __forceinline__ unsigned lds_offset(const void* p) {
  return (unsigned)(uintptr_t)p;
}

// Split 16 fp32 values into bf16 hi + bf16 residual lo fragments.
__device__ __forceinline__ void split_bf16(const float* v, v16bf& hi, v16bf& lo) {
#pragma unroll
  for (int e = 0; e < 16; ++e) {
    __bf16 h = (__bf16)v[e];
    hi[e] = h;
    lo[e] = (__bf16)(v[e] - (float)h);
  }
}

// Issue one TDM 2D tile load: `rows` rows of 32 fp32, row stride D_IN floats,
// into LDS at lds_byte_off with a 4-dword pad after every 32 dwords.
// D# layout per CDNA5 ISA ch. 8 (group0: 4 dwords, group1: 8 dwords).
__device__ __forceinline__ void tdm_load_tile(unsigned lds_byte_off,
                                              const float* gsrc, int rows) {
  unsigned long long ga = (unsigned long long)(uintptr_t)gsrc;
  v4u g0;
  g0[0] = 1u;                                        // count=1, user mode
  g0[1] = lds_byte_off;                              // lds_addr [63:32]
  g0[2] = (unsigned)ga;                              // global_addr [95:64]
  g0[3] = (unsigned)((ga >> 32) & 0x01ffffffu)       // global_addr [120:96]
        | (2u << 30);                                // type=2 ("image")
  v8i_t g1;
  g1[0] = (int)((2u << 16)        // data_size = 4 bytes
              | (1u << 20)        // pad_enable
              | (4u << 22)        // pad_interval: 2^(4+1)=32 dwords
              | (3u << 25));      // pad_amount: 4 dwords
  g1[1] = 0;                      // atomic_barrier=0; tensor_dim0[15:0]=0
  g1[2] = (int)0x00000100u;       // tensor_dim0 = 1<<24 (huge, no OOB clamp)
  g1[3] = (int)(0x00000100u | (32u << 16)); // tensor_dim1 = 1<<24; tile_dim0=32
  g1[4] = rows;                   // tile_dim1 = rows; tile_dim2 = 0
  g1[5] = (int)D_IN;              // tensor_dim0_stride = 4096 elements
  g1[6] = 0;                      // stride hi / dim1_stride lo
  g1[7] = 0;
  v4i_t gz = {0, 0, 0, 0};        // groups 2/3 unused (2D tile)
#if defined(__clang_major__) && (__clang_major__ >= 23)
  v8i_t gz8 = {0, 0, 0, 0, 0, 0, 0, 0};
  __builtin_amdgcn_tensor_load_to_lds(g0, g1, gz, gz, gz8, 0);
#else
  __builtin_amdgcn_tensor_load_to_lds(g0, g1, gz, gz, 0);
#endif
}

__global__ __launch_bounds__(256) void feature_gemm_kernel(
    const float* __restrict__ x, const float* __restrict__ w1,
    const float* __restrict__ b1, float* __restrict__ feature) {
  __shared__ float xbuf[2][XROWS * PADW];   // 2 x 18 KB
  __shared__ float wbuf[2][WROWS * PADW];   // 2 x  9 KB

  const int lane = threadIdx.x & 31;
  const int wave = threadIdx.x >> 5;
  const int r    = lane & 15;   // A: M row within tile / B: N col within tile
  const int half = lane >> 4;   // lane half selects K = 16*half..16*half+15 (contiguous)

  const int m0 = blockIdx.y * XROWS + wave * 16;
  const int n0 = blockIdx.x * WROWS;
  const bool issuer = (threadIdx.x < 32);   // wave 0 drives the TDM

  const float* xg = x  + (size_t)blockIdx.y * XROWS * D_IN;
  const float* wg = w1 + (size_t)n0 * D_IN;

  v8f acc[4];
#pragma unroll
  for (int t = 0; t < 4; ++t)
#pragma unroll
    for (int e = 0; e < 8; ++e) acc[t][e] = 0.0f;

  if (issuer) {
    tdm_load_tile(lds_offset(&xbuf[0][0]), xg, XROWS);
    tdm_load_tile(lds_offset(&wbuf[0][0]), wg, WROWS);
  }

  int cur = 0;
  for (int k = 0; k < D_IN; k += KSTEP) {
    if (issuer) {
      if (k + KSTEP < D_IN) {
        // Prefetch next K-slab into the other buffer, then retire the
        // two oldest TDM ops (current buffer) — TENSORcnt is in-order.
        tdm_load_tile(lds_offset(&xbuf[cur ^ 1][0]), xg + k + KSTEP, XROWS);
        tdm_load_tile(lds_offset(&wbuf[cur ^ 1][0]), wg + k + KSTEP, WROWS);
        __builtin_amdgcn_s_wait_tensorcnt(2);
      } else {
        __builtin_amdgcn_s_wait_tensorcnt(0);
      }
    }
    __syncthreads();   // publish TDM-written LDS tiles to all 8 waves

    // ---- A fragment (16x32 bf16) from LDS: lanes 0-15 hold K = 0..15,
    //      lanes 16-31 hold K = 16..31.
    const float* xr = &xbuf[cur][(wave * 16 + r) * PADW];
    float av[16];
    st4f(av + 0,  ld4(xr + 16 * half));
    st4f(av + 4,  ld4(xr + 16 * half + 4));
    st4f(av + 8,  ld4(xr + 16 * half + 8));
    st4f(av + 12, ld4(xr + 16 * half + 12));
    v16bf a_hi, a_lo;
    split_bf16(av, a_hi, a_lo);

#pragma unroll
    for (int nt = 0; nt < 4; ++nt) {
      // ---- B fragment (32x16 bf16) from LDS (tile shared by all waves):
      //      lane col n = r; lanes 0-15 hold K=0..15, lanes 16-31 K=16..31.
      const float* br = &wbuf[cur][(nt * 16 + r) * PADW + 16 * half];
      float bv[16];
      st4f(bv + 0,  ld4(br));
      st4f(bv + 4,  ld4(br + 4));
      st4f(bv + 8,  ld4(br + 8));
      st4f(bv + 12, ld4(br + 12));
      v16bf b_hi, b_lo;
      split_bf16(bv, b_hi, b_lo);

      acc[nt] = __builtin_amdgcn_wmma_f32_16x16x32_bf16(
          false, a_hi, false, b_hi, (short)0, acc[nt], false, false);
      acc[nt] = __builtin_amdgcn_wmma_f32_16x16x32_bf16(
          false, a_hi, false, b_lo, (short)0, acc[nt], false, false);
      acc[nt] = __builtin_amdgcn_wmma_f32_16x16x32_bf16(
          false, a_lo, false, b_hi, (short)0, acc[nt], false, false);
    }

    __syncthreads();   // all waves done reading buf[cur] before it is refilled
    cur ^= 1;
  }

  // Epilogue: bias + ReLU, store via C/D layout (VGPR v -> M = v + 8*half).
#pragma unroll
  for (int nt = 0; nt < 4; ++nt) {
    const int col  = n0 + nt * 16 + r;
    const float bias = b1[col];
#pragma unroll
    for (int v = 0; v < 8; ++v) {
      const int row = m0 + v + 8 * half;
      feature[(size_t)row * H1 + col] = fmaxf(acc[nt][v] + bias, 0.0f);
    }
  }
}

// ---------------------------------------------------------------------------
// Kernel 2: sequential RLS over 2048 rows; persistent grid with SW barrier.
// ---------------------------------------------------------------------------

#define RLS_NB  64
#define RLS_TPB 256

__device__ __forceinline__ void grid_barrier(int* cnt, int* gen, int nblocks) {
  __threadfence();          // make this thread's global writes device-visible
  __syncthreads();
  if (threadIdx.x == 0) {
    int g = __hip_atomic_load(gen, __ATOMIC_RELAXED, __HIP_MEMORY_SCOPE_AGENT);
    int a = __hip_atomic_fetch_add(cnt, 1, __ATOMIC_ACQ_REL, __HIP_MEMORY_SCOPE_AGENT);
    if (a == nblocks - 1) {
      __hip_atomic_store(cnt, 0, __ATOMIC_RELAXED, __HIP_MEMORY_SCOPE_AGENT);
      __hip_atomic_fetch_add(gen, 1, __ATOMIC_RELEASE, __HIP_MEMORY_SCOPE_AGENT);
    } else {
      while (__hip_atomic_load(gen, __ATOMIC_ACQUIRE, __HIP_MEMORY_SCOPE_AGENT) == g)
        __builtin_amdgcn_s_sleep(8);
    }
  }
  __syncthreads();
}

__global__ __launch_bounds__(RLS_TPB) void rls_kernel(
    const float* __restrict__ feature, const float* __restrict__ y,
    float* __restrict__ W, float* __restrict__ P,
    float* __restrict__ px, int* bar) {
  __shared__ float s_xi[HP];
  __shared__ float s_px[HP];
  const int tid  = threadIdx.x;
  const int gtid = blockIdx.x * RLS_TPB + tid;
  const int T    = RLS_NB * RLS_TPB;
  int* cnt = bar;
  int* gen = bar + 1;

  for (int i = 0; i < N_ROWS; ++i) {
    // xi = [feature[i, :], 1.0]  (Phi row)
    const float* frow = feature + (size_t)i * H1;
    for (int k = tid; k < HP; k += RLS_TPB) s_xi[k] = (k < H1) ? frow[k] : 1.0f;
    __syncthreads();

    // px[j] = sum_k P[j,k]*xi[k]. P stays exactly symmetric under the RLS
    // update, so we read column j (coalesced across lanes): P[k,j] == P[j,k].
    for (int j = gtid; j < HP; j += T) {
      float a = 0.0f;
      for (int k = 0; k < HP; ++k) a = fmaf(P[(size_t)k * HP + j], s_xi[k], a);
      px[j] = a;
    }
    grid_barrier(cnt, gen, RLS_NB);

    for (int k = tid; k < HP; k += RLS_TPB) s_px[k] = px[k];
    __syncthreads();

    // denom = 1 + xi.px (redundant per-thread; operands live in LDS)
    float denom = 1.0f;
    for (int k = 0; k < HP; ++k) denom = fmaf(s_xi[k], s_px[k], denom);
    const float inv = 1.0f / denom;

    // W update (block 0 only): W[j,o] -= px[j]*(xi@W - y[i])[o] / denom
    if (blockIdx.x == 0) {
      const int o = tid & 15;
      float e = -y[(size_t)i * D_OUT + o];
      for (int k = 0; k < HP; ++k) e = fmaf(s_xi[k], W[(size_t)k * D_OUT + o], e);
      __syncthreads();  // everyone reads old W before anyone writes
      for (int j = (tid >> 4); j < HP; j += RLS_TPB / 16)
        W[(size_t)j * D_OUT + o] -= s_px[j] * e * inv;
    }

    // P update (all blocks, row-partitioned): P[j,k] -= px[j]*px[k]/denom
    // (xi@P == px by symmetry, see above)
    for (int j = blockIdx.x; j < HP; j += RLS_NB) {
      const float pj = s_px[j] * inv;
      float* Prow = P + (size_t)j * HP;
      for (int k = tid; k < HP; k += RLS_TPB)
        Prow[k] = fmaf(-pj, s_px[k], Prow[k]);
    }
    grid_barrier(cnt, gen, RLS_NB);  // P consistent before next step's matvec
  }
}

// ---------------------------------------------------------------------------
// Kernel 3: out = feature @ W[:-1] + W[-1]   (fp32, tiny)
// ---------------------------------------------------------------------------

__global__ __launch_bounds__(256) void final_layer_kernel(
    const float* __restrict__ feature, const float* __restrict__ W,
    float* __restrict__ out) {
  const int idx = blockIdx.x * 256 + threadIdx.x;
  if (idx >= N_ROWS * D_OUT) return;
  const int o = idx & 15;
  const int n = idx >> 4;
  const float* f = feature + (size_t)n * H1;
  float acc = W[(size_t)H1 * D_OUT + o];  // bias row
  for (int h = 0; h < H1; ++h) acc = fmaf(f[h], W[(size_t)h * D_OUT + o], acc);
  out[idx] = acc;
}

__global__ void init_bar_kernel(int* bar) {
  if (threadIdx.x < 2) bar[threadIdx.x] = 0;
}

// ---------------------------------------------------------------------------

extern "C" void kernel_launch(void* const* d_in, const int* in_sizes, int n_in,
                              void* d_out, int out_size, void* d_ws, size_t ws_size,
                              hipStream_t stream) {
  const float* x  = (const float*)d_in[0];
  const float* y  = (const float*)d_in[1];
  const float* W1 = (const float*)d_in[2];
  const float* b1 = (const float*)d_in[3];
  const float* W0 = (const float*)d_in[4];
  const float* P0 = (const float*)d_in[5];
  float* out = (float*)d_out;

  // Workspace layout (256B-aligned slabs): ~12.7 MB total.
  char* ws = (char*)d_ws;
  float* feature = (float*)(ws);                                   // 8,388,608 B
  float* P       = (float*)(ws + 8388608);                         // 4,202,752 B
  float* W       = (float*)(ws + 8388608 + 4202752);               //    65,792 B
  float* px      = (float*)(ws + 8388608 + 4202752 + 65792);       //     4,352 B
  int*   bar     = (int*)  (ws + 8388608 + 4202752 + 65792 + 4352);

  // Working copies of the RLS state (inputs must not be mutated).
  hipMemcpyAsync(P, P0, (size_t)HP * HP * sizeof(float),
                 hipMemcpyDeviceToDevice, stream);
  hipMemcpyAsync(W, W0, (size_t)HP * D_OUT * sizeof(float),
                 hipMemcpyDeviceToDevice, stream);
  init_bar_kernel<<<1, 32, 0, stream>>>(bar);

  feature_gemm_kernel<<<dim3(H1 / WROWS, N_ROWS / XROWS), 256, 0, stream>>>(
      x, W1, b1, feature);
  rls_kernel<<<RLS_NB, RLS_TPB, 0, stream>>>(feature, y, W, P, px, bar);
  final_layer_kernel<<<(N_ROWS * D_OUT + 255) / 256, 256, 0, stream>>>(
      feature, W, out);

  (void)in_sizes; (void)n_in; (void)out_size; (void)ws_size;
}
